// SelfAttention1D_35107062678166
// MI455X (gfx1250) — matrix operand order repaired and checked by reading, not compiled, not run on hardware
//
#include <hip/hip_runtime.h>

typedef __bf16 bf16_t;
typedef __attribute__((ext_vector_type(16))) __bf16 v16bf;
typedef __attribute__((ext_vector_type(8)))  float  v8f;

#define WMMA_BF16(a, b, c) \
  __builtin_amdgcn_wmma_f32_16x16x32_bf16(false, (a), false, (b), (short)0, (c), false, false)

// ---- fragment loaders (CDNA5 16-bit WMMA VGPR layouts) ----
// A (16 x K=32, M=lane%16): lane<16 holds K {0..7,16..23}; lane>=16 holds K {8..15,24..31}
template <typename P>
__device__ inline v16bf load_frag_A(const P* rowptr, int lane) {
  union { v16bf v; uint4 q[2]; } u;
  const char* p = (const char*)rowptr + ((lane >> 4) << 4);  // +16B upper half
  u.q[0] = *(const uint4*)(p);
  u.q[1] = *(const uint4*)(p + 32);
  return u.v;
}
// B (K=32 x 16, N=lane%16): lane<16 holds K {0..7,16..23}; lane>=16 holds K {8..15,24..31}
template <typename P>
__device__ inline v16bf load_frag_B(const P* colptr, int lane) {
  union { v16bf v; uint4 q[2]; } u;
  const char* p = (const char*)colptr + ((lane >> 4) << 4);  // +16B upper half
  u.q[0] = *(const uint4*)(p);
  u.q[1] = *(const uint4*)(p + 32);
  return u.v;
}

// ------------------------------------------------------------------
// Kernel 0: convert weights f32 -> bf16 (then resident in L2)
// ------------------------------------------------------------------
__global__ __launch_bounds__(256) void cvt_weights(
    const float* __restrict__ Wq, const float* __restrict__ Wk,
    const float* __restrict__ Wv,
    bf16_t* __restrict__ wq, bf16_t* __restrict__ wk, bf16_t* __restrict__ wv) {
  int i = blockIdx.x * blockDim.x + threadIdx.x;   // 65536 threads total
  if (i < 32 * 256) { wq[i] = (bf16_t)Wq[i]; wk[i] = (bf16_t)Wk[i]; }
  if (i < 256 * 256) { wv[i] = (bf16_t)Wv[i]; }
}

// ------------------------------------------------------------------
// Kernel 1: fused q/k/v projection. One block = (b, 64 columns of L).
// [Wq;Wk;Wv] as 320x256 GEMM against x tile (256 x 64).
// q,k stored TRANSPOSED (l-major, 32 channels contiguous) so the flash
// kernel can build WMMA fragments straight from global/L2.
// ------------------------------------------------------------------
__global__ __launch_bounds__(256) void proj_kernel(
    const float* __restrict__ x,
    const bf16_t* __restrict__ wq, const bf16_t* __restrict__ wk,
    const bf16_t* __restrict__ wv,
    const float* __restrict__ bq, const float* __restrict__ bk,
    const float* __restrict__ bv,
    bf16_t* __restrict__ qTo, bf16_t* __restrict__ kTo, bf16_t* __restrict__ vo) {
  __shared__ __align__(16) bf16_t xT[64][264];   // [l][c], padded rows
  __shared__ float bias[320];

  const int b = blockIdx.y;
  const int l0 = blockIdx.x * 64;
  const int tid = threadIdx.x;
  const int lane = tid & 31;
  const int wave = tid >> 5;

  for (int i = tid; i < 320; i += 256) {
    float v;
    if (i < 32)       v = bq[i];
    else if (i < 64)  v = bk[i - 32];
    else              v = bv[i - 64];
    bias[i] = v;
  }
  const float* xb = x + (size_t)b * 256 * 4096 + l0;
  for (int i = tid; i < 256 * 64; i += 256) {
    int c = i >> 6, l = i & 63;
    xT[l][c] = (bf16_t)xb[(size_t)c * 4096 + l];
  }
  __syncthreads();

  // 20 row-tiles (320 out channels) x 4 col-tiles = 80 tiles over 8 waves
  for (int t = wave; t < 80; t += 8) {
    const int rt = t >> 2;
    const int ct = t & 3;
    const int ro = rt * 16;

    const bf16_t* Wbase;
    int rl;
    if (rt < 2)      { Wbase = wq; rl = ro; }
    else if (rt < 4) { Wbase = wk; rl = ro - 32; }
    else             { Wbase = wv; rl = ro - 64; }

    const bf16_t* arow = Wbase + (size_t)(rl + (lane & 15)) * 256;
    const bf16_t* brow = &xT[ct * 16 + (lane & 15)][0];

    v8f acc = {};
#pragma unroll
    for (int kk = 0; kk < 256; kk += 32) {
      v16bf a = load_frag_A(arow + kk, lane);
      v16bf bb = load_frag_B(brow + kk, lane);
      acc = WMMA_BF16(a, bb, acc);
    }

    const int col = l0 + ct * 16 + (lane & 15);
    const int rh = (lane >> 4) << 3;
    if (rt < 4) {
      // q / k: transposed store, 8 contiguous bf16 -> one b128 store
      bf16_t* tb = ((rt < 2) ? qTo : kTo) + (size_t)b * 4096 * 32 + (size_t)col * 32;
      union { uint4 q; __bf16 h[8]; } pk;
#pragma unroll
      for (int j = 0; j < 8; ++j)
        pk.h[j] = (bf16_t)(acc[j] + bias[ro + rh + j]);
      *(uint4*)&tb[rl + rh] = pk.q;
    } else {
      bf16_t* ob = vo + (size_t)b * 256 * 4096;
#pragma unroll
      for (int j = 0; j < 8; ++j)
        ob[(size_t)(rl + rh + j) * 4096 + col] = (bf16_t)(acc[j] + bias[ro + rh + j]);
    }
  }
}

// ------------------------------------------------------------------
// Kernel 2: flash attention. Block = (b, 32 queries), 10 waves.
// waves 0,1: S = Q^T K (one 16-query tile each), online softmax,
//            P/alpha -> double-buffered LDS.
// waves 2..9: each owns 32 channels, O += V * P^T (8 WMMAs / 64-key step).
// One barrier per 64-key step.
// ------------------------------------------------------------------
__global__ __launch_bounds__(320) void flash_kernel(
    const bf16_t* __restrict__ qT, const bf16_t* __restrict__ kT,
    const bf16_t* __restrict__ v,
    const float* __restrict__ x, const float* __restrict__ gamma,
    float* __restrict__ out) {
  __shared__ __align__(16) bf16_t pT[2][32][72];  // [buf][query][64 keys + pad]
  __shared__ float s_alpha[2][32];
  __shared__ float s_lsum[32];

  const int b = blockIdx.y;
  const int l0 = blockIdx.x * 32;
  const int tid = threadIdx.x;
  const int lane = tid & 31;
  const int wave = tid >> 5;

  const bf16_t* qTb = qT + (size_t)b * 4096 * 32;
  const bf16_t* kTb = kT + (size_t)b * 4096 * 32;
  const bf16_t* vb  = v  + (size_t)b * 256 * 4096;

  // softmax-wave state (loaded by all waves with clamped qtile; only 0,1 use it)
  const int qt = (wave < 2) ? wave : 0;
  v16bf afragQ = load_frag_A(qTb + (size_t)(l0 + qt * 16 + (lane & 15)) * 32, lane);
  float mrow[8], lrow[8];
#pragma unroll
  for (int j = 0; j < 8; ++j) { mrow[j] = -1e30f; lrow[j] = 0.0f; }

  // O-wave state
  const int cw = (wave - 2) * 32;
  v8f o00 = {}, o01 = {}, o10 = {}, o11 = {};
  v16bf va0, va1, va2, va3;

  for (int it = 0; it < 64; ++it) {
    const int m0 = it * 64;
    const int buf = it & 1;

    if (wave >= 2) {
      // issue V fragment loads early; they overlap the softmax waves' work
      const bf16_t* vrow = vb + (size_t)(cw + (lane & 15)) * 4096 + m0;
      va0 = load_frag_A(vrow, lane);                          // ch 0..15, K 0..31
      va1 = load_frag_A(vrow + 32, lane);                     // ch 0..15, K 32..63
      va2 = load_frag_A(vrow + (size_t)16 * 4096, lane);      // ch 16..31, K 0..31
      va3 = load_frag_A(vrow + (size_t)16 * 4096 + 32, lane); // ch 16..31, K 32..63
    } else {
      // ---- S = Q^T K over 64 keys (4 sub-tiles), fragments from global ----
      v8f s0 = {}, s1 = {}, s2 = {}, s3 = {};
      s0 = WMMA_BF16(afragQ, load_frag_B(kTb + (size_t)(m0 +  0 + (lane & 15)) * 32, lane), s0);
      s1 = WMMA_BF16(afragQ, load_frag_B(kTb + (size_t)(m0 + 16 + (lane & 15)) * 32, lane), s1);
      s2 = WMMA_BF16(afragQ, load_frag_B(kTb + (size_t)(m0 + 32 + (lane & 15)) * 32, lane), s2);
      s3 = WMMA_BF16(afragQ, load_frag_B(kTb + (size_t)(m0 + 48 + (lane & 15)) * 32, lane), s3);

      const int qrow = qt * 16 + ((lane >= 16) ? 8 : 0);
      const int key = lane & 15;
#pragma unroll
      for (int j = 0; j < 8; ++j) {
        float t = fmaxf(fmaxf(s0[j], s1[j]), fmaxf(s2[j], s3[j]));
#pragma unroll
        for (int off = 1; off < 16; off <<= 1)
          t = fmaxf(t, __shfl_xor(t, off, 32));
        float mn = fmaxf(mrow[j], t);
        float al = __expf(mrow[j] - mn);
        float p0 = __expf(s0[j] - mn);
        float p1 = __expf(s1[j] - mn);
        float p2 = __expf(s2[j] - mn);
        float p3 = __expf(s3[j] - mn);
        float rs = (p0 + p1) + (p2 + p3);
#pragma unroll
        for (int off = 1; off < 16; off <<= 1)
          rs += __shfl_xor(rs, off, 32);
        lrow[j] = al * lrow[j] + rs;
        mrow[j] = mn;
        pT[buf][qrow + j][key]      = (bf16_t)p0;
        pT[buf][qrow + j][key + 16] = (bf16_t)p1;
        pT[buf][qrow + j][key + 32] = (bf16_t)p2;
        pT[buf][qrow + j][key + 48] = (bf16_t)p3;
        if (key == 0) s_alpha[buf][qrow + j] = al;
      }
    }
    __syncthreads();   // pT[buf]/alpha[buf] ready; also fences reuse of buf

    if (wave >= 2) {
      const float al0 = s_alpha[buf][lane & 15];
      const float al1 = s_alpha[buf][16 + (lane & 15)];
      v16bf bp00 = load_frag_B(&pT[buf][(lane & 15)][0],       lane); // qtile0, K 0..31
      v16bf bp01 = load_frag_B(&pT[buf][(lane & 15)][32],      lane); // qtile0, K 32..63
      v16bf bp10 = load_frag_B(&pT[buf][16 + (lane & 15)][0],  lane); // qtile1, K 0..31
      v16bf bp11 = load_frag_B(&pT[buf][16 + (lane & 15)][32], lane); // qtile1, K 32..63
#pragma unroll
      for (int j = 0; j < 8; ++j) {
        o00[j] *= al0; o01[j] *= al0;
        o10[j] *= al1; o11[j] *= al1;
      }
      o00 = WMMA_BF16(va0, bp00, o00);
      o00 = WMMA_BF16(va1, bp01, o00);
      o01 = WMMA_BF16(va2, bp00, o01);
      o01 = WMMA_BF16(va3, bp01, o01);
      o10 = WMMA_BF16(va0, bp10, o10);
      o10 = WMMA_BF16(va1, bp11, o10);
      o11 = WMMA_BF16(va2, bp10, o11);
      o11 = WMMA_BF16(va3, bp11, o11);
    }
  }

  if (wave < 2) {
    const int qrow = qt * 16 + ((lane >= 16) ? 8 : 0);
    if ((lane & 15) == 0) {
#pragma unroll
      for (int j = 0; j < 8; ++j) s_lsum[qrow + j] = lrow[j];
    }
  }
  __syncthreads();

  if (wave >= 2) {
    const float g = gamma[0];
    const float inv0 = 1.0f / s_lsum[lane & 15];
    const float inv1 = 1.0f / s_lsum[16 + (lane & 15)];
    const int col0 = l0 + (lane & 15);
    const int col1 = col0 + 16;
    const int rh = (lane >> 4) << 3;
    const float* xb = x + (size_t)b * 256 * 4096;
    float* ob = out + (size_t)b * 256 * 4096;
#pragma unroll
    for (int j = 0; j < 8; ++j) {
      size_t r0 = (size_t)(cw + rh + j) * 4096;
      size_t r1 = (size_t)(cw + 16 + rh + j) * 4096;
      ob[r0 + col0] = g * (o00[j] * inv0) + xb[r0 + col0];
      ob[r1 + col0] = g * (o01[j] * inv0) + xb[r1 + col0];
      ob[r0 + col1] = g * (o10[j] * inv1) + xb[r0 + col1];
      ob[r1 + col1] = g * (o11[j] * inv1) + xb[r1 + col1];
    }
  }
}

// ------------------------------------------------------------------
extern "C" void kernel_launch(void* const* d_in, const int* in_sizes, int n_in,
                              void* d_out, int out_size, void* d_ws, size_t ws_size,
                              hipStream_t stream) {
  const float* x     = (const float*)d_in[0];
  const float* Wq    = (const float*)d_in[1];
  const float* bq    = (const float*)d_in[2];
  const float* Wk    = (const float*)d_in[3];
  const float* bk    = (const float*)d_in[4];
  const float* Wv    = (const float*)d_in[5];
  const float* bv    = (const float*)d_in[6];
  const float* gamma = (const float*)d_in[7];
  float* out = (float*)d_out;

  char* ws = (char*)d_ws;
  bf16_t* wq_bf = (bf16_t*)ws;  ws += (size_t)32 * 256 * 2;        // 16 KB
  bf16_t* wk_bf = (bf16_t*)ws;  ws += (size_t)32 * 256 * 2;        // 16 KB
  bf16_t* wv_bf = (bf16_t*)ws;  ws += (size_t)256 * 256 * 2;       // 128 KB
  bf16_t* qT_ws = (bf16_t*)ws;  ws += (size_t)4 * 4096 * 32 * 2;   // 1 MB  [b][l][32]
  bf16_t* kT_ws = (bf16_t*)ws;  ws += (size_t)4 * 4096 * 32 * 2;   // 1 MB  [b][m][32]
  bf16_t* v_ws  = (bf16_t*)ws;  ws += (size_t)4 * 256 * 4096 * 2;  // 8 MB  [b][c][m]

  cvt_weights<<<256, 256, 0, stream>>>(Wq, Wk, Wv, wq_bf, wk_bf, wv_bf);

  proj_kernel<<<dim3(64, 4), 256, 0, stream>>>(
      x, wq_bf, wk_bf, wv_bf, bq, bk, bv, qT_ws, kT_ws, v_ws);

  flash_kernel<<<dim3(128, 4), 320, 0, stream>>>(
      qT_ws, kT_ws, v_ws, x, gamma, out);
}
